// MinimalBeatDecoder_3582002724939
// MI455X (gfx1250) — compile-verified, matrix operand order checked by reading, not run on hardware
//
#include <hip/hip_runtime.h>
#include <hip/hip_bf16.h>

#define NFRAMES   (1 << 24)
#define MAX_BEATS (NFRAMES / 8)     // 2^21
#define TPB       256
#define IPT       16
#define CHUNK     (TPB * IPT)       // 4096 elements per block
#define NBLK      (NFRAMES / CHUNK) // 4096 blocks
#define HALO      4                 // need x[i-4..i+4] beyond chunk for peakL/peakR

#define FLAG_AGG    1u
#define FLAG_PREFIX 2u
#define VAL_MASK    0x3FFFFFFFu

typedef __attribute__((ext_vector_type(16))) _Float16 v16h;
typedef __attribute__((ext_vector_type(8)))  float    v8f;

// ---------------------------------------------------------------------------
// Stage x[base-4 .. base+CHUNK+3] into lds[0..CHUNK+7].
// Bulk CHUNK floats via async global->LDS B128 copies (ASYNCcnt path),
// halos via bounds-checked scalar loads (-inf padding like lax.reduce_window).
// ---------------------------------------------------------------------------
__device__ __forceinline__ void stage_tile(const float* __restrict__ x,
                                           float* lds, int base, int tid) {
  const float* src = x + base;
  const unsigned long long sbase = (unsigned long long)(uintptr_t)src;
  const unsigned lbase = (unsigned)(uintptr_t)(&lds[HALO]);
#pragma unroll
  for (int r = 0; r < 4; ++r) {
    unsigned lds_off = lbase + (unsigned)((r * TPB + tid) * 16);
    unsigned goff    = (unsigned)((r * TPB + tid) * 16);
    // GVS form: vdst = LDS byte address, vaddr = 32-bit offset, saddr = base
    asm volatile("global_load_async_to_lds_b128 %0, %1, %2"
                 :: "v"(lds_off), "v"(goff), "s"(sbase)
                 : "memory");
  }
  if (tid < 2 * HALO) {
    int j, gi;
    if (tid < HALO) { j = tid;                         gi = base - HALO + tid;           }
    else            { j = CHUNK + HALO + (tid - HALO); gi = base + CHUNK + (tid - HALO); }
    float v = (gi >= 0 && gi < NFRAMES) ? x[gi] : -__builtin_inff();
    lds[j] = v;
  }
  asm volatile("s_wait_asynccnt 0" ::: "memory");
  __syncthreads();
}

// peak[p] for the 18 positions p = tid*16-1 .. tid*16+16 (chunk-local),
// then derive run-start / run-end bitmasks for this thread's 16 elements.
__device__ __forceinline__ void peak_masks(const float* lds, int tid,
                                           unsigned& isNew, unsigned& isEnd) {
  unsigned pk = 0;
  const int j0 = tid * IPT + HALO - 1; // lds index of position p0 = tid*16 - 1
#pragma unroll
  for (int k = 0; k < IPT + 2; ++k) {
    int j = j0 + k;
    float v = lds[j];
    float m = fmaxf(fmaxf(lds[j - 3], lds[j - 2]), lds[j - 1]);
    m = fmaxf(m, fmaxf(fmaxf(lds[j + 1], lds[j + 2]), lds[j + 3]));
    if (v >= m && v > 0.0f) pk |= (1u << k);
  }
  unsigned cur = (pk >> 1) & 0xFFFFu;  // peak at p
  unsigned prv =  pk       & 0xFFFFu;  // peak at p-1
  unsigned nxt = (pk >> 2) & 0xFFFFu;  // peak at p+1
  isNew = cur & ~prv;                  // gap>MERGE_INTERVAL(=1) <=> !peak[p-1]
  isEnd = cur & ~nxt;
}

// ---------------------------------------------------------------------------
// Reset decoupled-lookback state (ws is poisoned / stale between calls).
// ---------------------------------------------------------------------------
__global__ __launch_bounds__(TPB) void beat_init_kernel(unsigned* __restrict__ stateS,
                                                        unsigned* __restrict__ stateE) {
  int k = blockIdx.x * blockDim.x + threadIdx.x; // NBLK entries total
  stateS[k] = 0u;
  stateE[k] = 0u;
}

// ---------------------------------------------------------------------------
// Single pass over x: peaks -> WMMA block totals -> publish aggregate ->
// intra-block scan -> decoupled lookback -> scatter starts (ws) / ends (out).
// ---------------------------------------------------------------------------
__global__ __launch_bounds__(TPB) void beat_fused_kernel(
    const float* __restrict__ x,
    unsigned* __restrict__ stateS, unsigned* __restrict__ stateE,
    int* __restrict__ starts, float* __restrict__ outEnds) {
  __shared__ float lds[CHUNK + 2 * HALO];
  __shared__ __align__(32) _Float16 hcnt[2 * TPB];
  __shared__ int scn[2 * TPB];
  __shared__ int bcast[2];
  const int tid = threadIdx.x;
  const int blk = blockIdx.x;
  const int base = blk * CHUNK;

  stage_tile(x, lds, base, tid);
  unsigned isNew, isEnd;
  peak_masks(lds, tid, isNew, isEnd);

  const int cS = __popc(isNew), cE = __popc(isEnd);
  hcnt[tid]       = (_Float16)cS;
  hcnt[TPB + tid] = (_Float16)cE;
  scn[tid]        = cS;
  scn[TPB + tid]  = cE;
  __syncthreads();

  // --- WMMA block-total reduction (wave 0, EXEC all ones) -------------------
  // A(16x32 f16): rows 0-7 = 256 start-counts, rows 8-15 = 256 end-counts,
  // B = ones => D row-sums; lanes 0-15 get total(S), lanes 16-31 total(E).
  int itot = 0;
  if (tid < 32) {
    int grp, off;
    if      (tid <  8) { grp = tid;      off = 0;   }
    else if (tid < 16) { grp = tid - 8;  off = TPB; }
    else if (tid < 24) { grp = tid - 8;  off = 0;   }
    else               { grp = tid - 16; off = TPB; }
    v16h a = *(const v16h*)&hcnt[off + grp * 16];
    v16h b;
#pragma unroll
    for (int i = 0; i < 16; ++i) b[i] = (_Float16)1.0f;
    v8f c = {};
    c = __builtin_amdgcn_wmma_f32_16x16x32_f16(false, a, false, b,
                                               (short)0, c, false, false);
    float tot = ((c[0] + c[1]) + (c[2] + c[3])) + ((c[4] + c[5]) + (c[6] + c[7]));
    itot = (int)(tot + 0.5f);
    // Publish aggregate (or final prefix for block 0) as early as possible so
    // successors' lookback overlaps with our intra-block scan below.
    if (tid == 0 || tid == 16) {
      unsigned* st = (tid == 0) ? stateS : stateE;
      unsigned pub = ((blk == 0) ? (FLAG_PREFIX << 30) : (FLAG_AGG << 30))
                     | (unsigned)itot;
      __hip_atomic_store(&st[blk], pub, __ATOMIC_RELEASE, __HIP_MEMORY_SCOPE_AGENT);
    }
  }

  // --- intra-block Hillis-Steele scan over per-thread counts ---------------
  for (int d = 1; d < TPB; d <<= 1) {
    int a  = scn[tid],                  b  = scn[TPB + tid];
    int au = (tid >= d) ? scn[tid - d] : 0;
    int bu = (tid >= d) ? scn[TPB + tid - d] : 0;
    __syncthreads();
    scn[tid] = a + au;
    scn[TPB + tid] = b + bu;
    __syncthreads();
  }

  // --- decoupled lookback: lane 0 resolves S, lane 16 resolves E -----------
  if (tid == 0 || tid == 16) {
    unsigned* st = (tid == 0) ? stateS : stateE;
    int bse = 0;
    if (blk > 0) {
      int p = blk - 1;
      for (;;) {
        unsigned w = __hip_atomic_load(&st[p], __ATOMIC_ACQUIRE,
                                       __HIP_MEMORY_SCOPE_AGENT);
        unsigned f = w >> 30;
        if (f == 0u) { __builtin_amdgcn_s_sleep(1); continue; }
        bse += (int)(w & VAL_MASK);
        if (f >= FLAG_PREFIX) break;   // hit an inclusive prefix: done
        --p;                           // aggregate: keep walking back
      }
      __hip_atomic_store(&st[blk],
                         (FLAG_PREFIX << 30) | (unsigned)(bse + itot),
                         __ATOMIC_RELEASE, __HIP_MEMORY_SCOPE_AGENT);
    }
    bcast[(tid == 0) ? 0 : 1] = bse;
  }
  __syncthreads();

  // --- scatter ranked starts / ends -----------------------------------------
  int rS = bcast[0] + (tid ? scn[tid - 1] : 0);
  int rE = bcast[1] + (tid ? scn[TPB + tid - 1] : 0);
#pragma unroll
  for (int k = 0; k < IPT; ++k) {
    int i = base + tid * IPT + k;
    if ((isNew >> k) & 1u) { if (rS < MAX_BEATS) starts[rS] = i;         ++rS; }
    if ((isEnd >> k) & 1u) { if (rE < MAX_BEATS) outEnds[rE] = (float)i; ++rE; }
  }
}

// ---------------------------------------------------------------------------
// beats[k] = (start+end)/2 for k < nRuns, else -1. nRuns = last block's prefix.
// ---------------------------------------------------------------------------
__global__ __launch_bounds__(TPB) void beat_finalize_kernel(
    const int* __restrict__ starts, const unsigned* __restrict__ stateS,
    float* __restrict__ out) {
  int k = blockIdx.x * blockDim.x + threadIdx.x;
  int n = (int)(stateS[NBLK - 1] & VAL_MASK);
  n = (n > MAX_BEATS) ? MAX_BEATS : n;
  float e = out[k];                 // end index staged here by the fused pass
  out[k] = (k < n) ? 0.5f * ((float)starts[k] + e) : -1.0f;
}

extern "C" void kernel_launch(void* const* d_in, const int* in_sizes, int n_in,
                              void* d_out, int out_size, void* d_ws, size_t ws_size,
                              hipStream_t stream) {
  (void)in_sizes; (void)n_in; (void)out_size; (void)ws_size;
  const float* x   = (const float*)d_in[0];
  float*       out = (float*)d_out;
  char*        ws  = (char*)d_ws;

  unsigned* stateS = (unsigned*)(ws);           // NBLK u32 (16 KB)
  unsigned* stateE = (unsigned*)(ws + 16384);   // NBLK u32 (16 KB)
  int*      starts = (int*)(ws + 65792);        // MAX_BEATS ints (8 MB)

  beat_init_kernel<<<NBLK / TPB, TPB, 0, stream>>>(stateS, stateE);
  beat_fused_kernel<<<NBLK, TPB, 0, stream>>>(x, stateS, stateE, starts, out);
  beat_finalize_kernel<<<MAX_BEATS / TPB, TPB, 0, stream>>>(starts, stateS, out);
}
